// EncoderBlock_6193342841033
// MI455X (gfx1250) — matrix operand order repaired and checked by reading, not compiled, not run on hardware
//
#include <hip/hip_runtime.h>
#include <cstdint>

// ---------------------------------------------------------------------------
// Encoder block for MI455X (gfx1250, wave32, WMMA).
// - All GEMMs: v_wmma_f32_16x16x32_bf16, LDS-staged tiles, double-buffered
//   with global_load_async_to_lds_b128 (ASYNCcnt) when the builtin exists.
// - Attention: full 16x2048 fp32 score strip + bf16 P strip in LDS
//   (194KB/WG; CDNA5 allows 320KB), exact two-pass softmax.
// ---------------------------------------------------------------------------

#define DEV __device__ __forceinline__

#if defined(__has_builtin)
#  if __has_builtin(__builtin_amdgcn_global_load_async_to_lds_b128)
#    define HAVE_ASYNC_LDS 1
#  else
#    define HAVE_ASYNC_LDS 0
#  endif
#  if __has_builtin(__builtin_amdgcn_s_wait_asynccnt)
#    define HAVE_WAIT_ASYNC 1
#  else
#    define HAVE_WAIT_ASYNC 0
#  endif
#else
#  define HAVE_ASYNC_LDS 0
#  define HAVE_WAIT_ASYNC 0
#endif

typedef __attribute__((ext_vector_type(16))) __bf16 v16bf;
typedef __attribute__((ext_vector_type(16))) unsigned short v16u;
typedef __attribute__((ext_vector_type(8)))  float v8f;
typedef __attribute__((ext_vector_type(4)))  int   v4i;

static constexpr int CB   = 2;
static constexpr int CS   = 2048;
static constexpr int CD   = 1024;
static constexpr int CH   = 16;
static constexpr int CDK  = 64;
static constexpr int CDFF = 4096;
static constexpr int CMT  = CB * CS;           // 4096 tokens
static constexpr float CEPS = 1.1920928955078125e-07f;

struct Frag32B { uint4 lo, hi; };

DEV unsigned short f2bf(float f) {
  union { float f; unsigned u; } v; v.f = f;
  unsigned r = v.u + 0x7FFFu + ((v.u >> 16) & 1u);
  return (unsigned short)(r >> 16);
}
DEV float bf2f(unsigned short b) {
  union { unsigned u; float f; } v; v.u = ((unsigned)b) << 16; return v.f;
}
DEV v16bf load_frag(const unsigned short* p0, const unsigned short* p1) {
  Frag32B t;
  t.lo = *(const uint4*)p0;
  t.hi = *(const uint4*)p1;
  return __builtin_bit_cast(v16bf, t);
}

#if HAVE_ASYNC_LDS
// 16B per-lane async copy global -> LDS (tracked by ASYNCcnt).
// Signature (from hipcc diagnostic): (v4i __device__*, v4i __shared__*, int, int)
DEV void async_copy16(const void* g, void* l) {
  __builtin_amdgcn_global_load_async_to_lds_b128(
      (__attribute__((address_space(1))) v4i*)(uintptr_t)g,
      (__attribute__((address_space(3))) v4i*)(uint32_t)(uintptr_t)l,
      0, 0);
}
DEV void wait_async0() {
#if HAVE_WAIT_ASYNC
  __builtin_amdgcn_s_wait_asynccnt(0);
#else
  asm volatile("s_wait_asynccnt 0x0" ::: "memory");
#endif
}
#endif

// ---------------------------------------------------------------------------
// RMSNorm: x (f32, rows x d) -> out (bf16), scaled by g.
// ---------------------------------------------------------------------------
__global__ __launch_bounds__(256)
void rmsnorm_kernel(const float* __restrict__ x, const float* __restrict__ g,
                    unsigned short* __restrict__ out, int d) {
  __shared__ float red[256];
  const int row = blockIdx.x;
  const float* xr = x + (size_t)row * d;
  float ss = 0.0f;
  for (int i = threadIdx.x; i < d; i += 256) { float v = xr[i]; ss += v * v; }
  red[threadIdx.x] = ss;
  __syncthreads();
  for (int s = 128; s > 0; s >>= 1) {
    if ((int)threadIdx.x < s) red[threadIdx.x] += red[threadIdx.x + s];
    __syncthreads();
  }
  const float scale = rsqrtf(red[0] / (float)d + CEPS);
  for (int i = threadIdx.x; i < d; i += 256)
    out[(size_t)row * d + i] = f2bf(xr[i] * scale * g[i]);
}

// ---------------------------------------------------------------------------
// Convert f32 [K][N] weight -> bf16 transposed [N][K].
// ---------------------------------------------------------------------------
__global__ __launch_bounds__(256)
void cvt_t_kernel(const float* __restrict__ in, unsigned short* __restrict__ out,
                  int K, int N) {
  size_t idx = (size_t)blockIdx.x * 256 + threadIdx.x;
  if (idx >= (size_t)K * N) return;
  int k = (int)(idx / N);
  int n = (int)(idx % N);
  out[(size_t)n * K + k] = f2bf(in[idx]);
}

// ---------------------------------------------------------------------------
// WMMA GEMM: C[M,N] = A[M,K](bf16) @ Bt[N,K](bf16)^T + bias.
// 256 threads / 8 waves, 64x64 tile. A and B 64x32 tiles staged in LDS,
// double-buffered via async global->LDS loads.
// ---------------------------------------------------------------------------
template<bool RELU, bool RESID, bool OUT_F32, bool OUT_BF16, bool OUT_VT>
__global__ __launch_bounds__(256)
void gemm_kernel(const unsigned short* __restrict__ A,
                 const unsigned short* __restrict__ Bt,
                 const float* __restrict__ bias,
                 const float* __restrict__ resid,
                 float* __restrict__ outf,
                 unsigned short* __restrict__ outb,
                 int M, int N, int K) {
  __shared__ unsigned short s_a[2][64 * 32];
  __shared__ unsigned short s_b[2][64 * 32];

  const int tid  = threadIdx.x;
  const int lane = tid & 31;
  const int wave = tid >> 5;
  const int tile_m = blockIdx.y << 6;
  const int tile_n = blockIdx.x << 6;
  const int ni  = wave & 3;
  const int m0  = (wave >> 2) << 1;
  const int l15 = lane & 15;
  const int lhi = lane >> 4;

  // Staging: thread t copies 8 halves of row (t>>2), chunk (t&3).
  const int sr = tid >> 2;
  const int sc = (tid & 3) << 3;
  const unsigned short* gA = A  + (size_t)(tile_m + sr) * K + sc;
  const unsigned short* gB = Bt + (size_t)(tile_n + sr) * K + sc;
  const int soff = sr * 32 + sc;

  auto stage = [&](int buf, int kk) {
#if HAVE_ASYNC_LDS
    async_copy16(gA + kk, &s_a[buf][soff]);
    async_copy16(gB + kk, &s_b[buf][soff]);
#else
    *(uint4*)&s_a[buf][soff] = *(const uint4*)(gA + kk);
    *(uint4*)&s_b[buf][soff] = *(const uint4*)(gB + kk);
#endif
  };

  v8f acc0 = {};
  v8f acc1 = {};
  stage(0, 0);
  int cur = 0;
  for (int kk = 0; kk < K; kk += 32) {
#if HAVE_ASYNC_LDS
    wait_async0();
#endif
    // One barrier: current buffer filled AND previous readers of the
    // other buffer are done before we refill it below.
    __syncthreads();
    if (kk + 32 < K) stage(cur ^ 1, kk + 32);

    const unsigned short* ha0 = &s_a[cur][(m0 * 16 + l15) * 32 + (lhi << 3)];
    const unsigned short* ha1 = ha0 + 16 * 32;
    const unsigned short* hb  = &s_b[cur][(ni * 16 + l15) * 32 + (lhi << 3)];
    v16bf a0 = load_frag(ha0, ha0 + 16);
    v16bf a1 = load_frag(ha1, ha1 + 16);
    v16bf b  = load_frag(hb,  hb + 16);
    acc0 = __builtin_amdgcn_wmma_f32_16x16x32_bf16(false, a0, false, b,
                                                   (short)0, acc0, false, false);
    acc1 = __builtin_amdgcn_wmma_f32_16x16x32_bf16(false, a1, false, b,
                                                   (short)0, acc1, false, false);
    cur ^= 1;
  }

  const int n  = tile_n + ni * 16 + l15;
  const float bv = bias[n];
#pragma unroll
  for (int half = 0; half < 2; ++half) {
#pragma unroll
    for (int j = 0; j < 8; ++j) {
      const int m = tile_m + (m0 + half) * 16 + j + (lhi << 3);
      float v = (half ? acc1[j] : acc0[j]) + bv;
      if constexpr (RELU) v = fmaxf(v, 0.0f);
      if constexpr (RESID) v += resid[(size_t)m * N + n];
      if constexpr (OUT_F32)  outf[(size_t)m * N + n] = v;
      if constexpr (OUT_BF16) outb[(size_t)m * N + n] = f2bf(v);
      if constexpr (OUT_VT) {
        const int bb = m >> 11, s = m & (CS - 1);
        const int hh = n >> 6,  dk = n & (CDK - 1);
        outb[((((size_t)bb * CH + hh) * CDK + dk) << 11) + s] = f2bf(v);
      }
    }
  }
}

// ---------------------------------------------------------------------------
// RoPE applied in place on bf16 q and k (token-major [MT][D]).
// ---------------------------------------------------------------------------
__global__ __launch_bounds__(256)
void rope_kernel(unsigned short* __restrict__ q, unsigned short* __restrict__ k) {
  const unsigned idx = blockIdx.x * 256u + threadIdx.x;   // < MT * 512
  const int token = idx >> 9;
  const int p = idx & 511;
  const int h = p >> 5;
  const int i = p & 31;
  const int s = token & (CS - 1);
  const float freq = exp2f(-(float)i * (13.287712379549449f / 32.0f));
  const float ang = (float)s * freq;
  const float c = cosf(ang), sn = sinf(ang);
  const size_t base = (size_t)token * CD + h * CDK;
  {
    float x1 = bf2f(q[base + i]), x2 = bf2f(q[base + i + 32]);
    q[base + i]      = f2bf(x1 * c - x2 * sn);
    q[base + i + 32] = f2bf(x2 * c + x1 * sn);
  }
  {
    float x1 = bf2f(k[base + i]), x2 = bf2f(k[base + i + 32]);
    k[base + i]      = f2bf(x1 * c - x2 * sn);
    k[base + i + 32] = f2bf(x2 * c + x1 * sn);
  }
}

// ---------------------------------------------------------------------------
// Attention: one block = one (b, h, 16-query tile). 128 threads = 4 waves.
// LDS: 16x2048 fp32 scores (128KB) + 16x2048 bf16 P (64KB) + 16x64 q (2KB).
// ---------------------------------------------------------------------------
__global__ __launch_bounds__(128)
void attn_kernel(const unsigned short* __restrict__ qb,
                 const unsigned short* __restrict__ kb,
                 const unsigned short* __restrict__ vt,
                 unsigned short* __restrict__ ob) {
  extern __shared__ char smem[];
  float* s_scores = (float*)smem;                                   // 16 x CS
  unsigned short* s_p = (unsigned short*)(smem + 16 * CS * 4);      // 16 x CS
  unsigned short* s_q = (unsigned short*)(smem + 16 * CS * 4 + 16 * CS * 2);

  const int blk = blockIdx.x;
  const int qt = blk & 127;
  const int h  = (blk >> 7) & (CH - 1);
  const int b  = blk >> 11;
  const int tid = threadIdx.x;
  const int lane = tid & 31;
  const int wave = tid >> 5;
  const int l15 = lane & 15;
  const int lhi = lane >> 4;
  const size_t qrow0 = (size_t)b * CS + qt * 16;

  __shared__ float red[128];
  __shared__ float rowmax[16];
  __shared__ float rowinv[16];

  // Load the 16x64 bf16 q tile (128 threads x 16B).
  {
    const int r = tid >> 3;
    const int c8 = (tid & 7) << 3;
#if HAVE_ASYNC_LDS
    async_copy16(qb + (qrow0 + r) * CD + h * CDK + c8, s_q + r * CDK + c8);
    wait_async0();
#else
    *(uint4*)(s_q + r * CDK + c8) =
        *(const uint4*)(qb + (qrow0 + r) * CD + h * CDK + c8);
#endif
  }
  __syncthreads();

  // ---- scores = (q @ k^T) / sqrt(DK) ----
  const float scale = 0.125f;
  for (int kt = wave; kt < CS / 16; kt += 4) {
    const size_t keyrow = (size_t)b * CS + kt * 16 + l15;
    v8f acc = {};
#pragma unroll
    for (int kk = 0; kk < 2; ++kk) {
      const unsigned short* ha = s_q + l15 * CDK + kk * 32 + (lhi << 3);
      const unsigned short* hb = kb + keyrow * CD + h * CDK + kk * 32 + (lhi << 3);
      v16bf a = load_frag(ha, ha + 16);
      v16bf bfr = load_frag(hb, hb + 16);
      acc = __builtin_amdgcn_wmma_f32_16x16x32_bf16(false, a, false, bfr,
                                                    (short)0, acc, false, false);
    }
#pragma unroll
    for (int j = 0; j < 8; ++j) {
      const int m = j + (lhi << 3);
      s_scores[m * CS + kt * 16 + l15] = acc[j] * scale;
    }
  }
  __syncthreads();

  // ---- softmax pass 1: row max ----
  {
    const int r = tid & 15, c = tid >> 4;
    float mx = -3.402823466e38f;
    const float* row = s_scores + r * CS + c * 256;
    for (int i = 0; i < 256; ++i) mx = fmaxf(mx, row[i]);
    red[tid] = mx;
  }
  __syncthreads();
  if (tid < 16) {
    float mx = red[tid];
    for (int c = 1; c < 8; ++c) mx = fmaxf(mx, red[tid + c * 16]);
    rowmax[tid] = mx;
  }
  __syncthreads();

  // ---- softmax pass 2: exp -> bf16 P strip, accumulate row sums ----
  {
    const int r = tid & 15, c = tid >> 4;
    const float mx = rowmax[r];
    float sum = 0.0f;
    const float* row = s_scores + r * CS + c * 256;
    unsigned short* prow = s_p + r * CS + c * 256;
    for (int i = 0; i < 256; ++i) {
      float e = __expf(row[i] - mx);
      prow[i] = f2bf(e);
      sum += e;
    }
    red[tid] = sum;
  }
  __syncthreads();
  if (tid < 16) {
    float s = 0.0f;
    for (int c = 0; c < 8; ++c) s += red[tid + c * 16];
    rowinv[tid] = 1.0f / s;
  }
  __syncthreads();

  // ---- o = P @ V : wave w owns dk columns [w*16, w*16+16) ----
  {
    v8f acc = {};
    const unsigned short* vbase =
        vt + (((size_t)b * CH + h) * CDK + wave * 16 + l15) * CS;
    for (int ks = 0; ks < CS / 32; ++ks) {
      const unsigned short* ha = s_p + l15 * CS + ks * 32 + (lhi << 3);
      const unsigned short* hb = vbase + ks * 32 + (lhi << 3);
      v16bf a = load_frag(ha, ha + 16);
      v16bf bfr = load_frag(hb, hb + 16);
      acc = __builtin_amdgcn_wmma_f32_16x16x32_bf16(false, a, false, bfr,
                                                    (short)0, acc, false, false);
    }
#pragma unroll
    for (int j = 0; j < 8; ++j) {
      const int m = j + (lhi << 3);
      const float v = acc[j] * rowinv[m];
      ob[(qrow0 + m) * CD + h * CDK + wave * 16 + l15] = f2bf(v);
    }
  }
}

// ---------------------------------------------------------------------------
// Host-side orchestration.
// ---------------------------------------------------------------------------
extern "C" void kernel_launch(void* const* d_in, const int* in_sizes, int n_in,
                              void* d_out, int out_size, void* d_ws, size_t ws_size,
                              hipStream_t stream) {
  const float* x  = (const float*)d_in[0];
  const float* wq = (const float*)d_in[1];
  const float* bq = (const float*)d_in[2];
  const float* wk = (const float*)d_in[3];
  const float* bk = (const float*)d_in[4];
  const float* wv = (const float*)d_in[5];
  const float* bv = (const float*)d_in[6];
  const float* wo = (const float*)d_in[7];
  const float* bo = (const float*)d_in[8];
  const float* w1 = (const float*)d_in[9];
  const float* b1 = (const float*)d_in[10];
  const float* w2 = (const float*)d_in[11];
  const float* b2 = (const float*)d_in[12];
  const float* g1 = (const float*)d_in[13];
  const float* g2 = (const float*)d_in[14];
  float* out = (float*)d_out;

  char* ws = (char*)d_ws;
  size_t off = 0;
  auto alloc = [&](size_t bytes) -> void* {
    void* p = ws + off;
    off += (bytes + 255) & ~(size_t)255;
    return p;
  };
  unsigned short* h_bf  = (unsigned short*)alloc((size_t)CMT * CD * 2);
  unsigned short* wqt   = (unsigned short*)alloc((size_t)CD * CD * 2);
  unsigned short* wkt   = (unsigned short*)alloc((size_t)CD * CD * 2);
  unsigned short* wvt   = (unsigned short*)alloc((size_t)CD * CD * 2);
  unsigned short* wot   = (unsigned short*)alloc((size_t)CD * CD * 2);
  unsigned short* w1t   = (unsigned short*)alloc((size_t)CDFF * CD * 2);
  unsigned short* w2t   = (unsigned short*)alloc((size_t)CD * CDFF * 2);
  unsigned short* q_bf  = (unsigned short*)alloc((size_t)CMT * CD * 2);
  unsigned short* k_bf  = (unsigned short*)alloc((size_t)CMT * CD * 2);
  unsigned short* vt    = (unsigned short*)alloc((size_t)CMT * CD * 2);
  unsigned short* o_bf  = (unsigned short*)alloc((size_t)CMT * CD * 2);
  float*          x2    = (float*)alloc((size_t)CMT * CD * 4);
  unsigned short* h2_bf = (unsigned short*)alloc((size_t)CMT * CD * 2);
  unsigned short* a1_bf = (unsigned short*)alloc((size_t)CMT * CDFF * 2);

  rmsnorm_kernel<<<CMT, 256, 0, stream>>>(x, g1, h_bf, CD);

  const int gDD = (CD * CD) / 256;
  const int gDF = (CD * CDFF) / 256;
  cvt_t_kernel<<<gDD, 256, 0, stream>>>(wq, wqt, CD, CD);
  cvt_t_kernel<<<gDD, 256, 0, stream>>>(wk, wkt, CD, CD);
  cvt_t_kernel<<<gDD, 256, 0, stream>>>(wv, wvt, CD, CD);
  cvt_t_kernel<<<gDD, 256, 0, stream>>>(wo, wot, CD, CD);
  cvt_t_kernel<<<gDF, 256, 0, stream>>>(w1, w1t, CD, CDFF);
  cvt_t_kernel<<<gDF, 256, 0, stream>>>(w2, w2t, CDFF, CD);

  dim3 gD(CD / 64, CMT / 64);      // (16, 64)
  dim3 gF(CDFF / 64, CMT / 64);    // (64, 64)
  gemm_kernel<false, false, false, true, false><<<gD, 256, 0, stream>>>(
      h_bf, wqt, bq, nullptr, nullptr, q_bf, CMT, CD, CD);
  gemm_kernel<false, false, false, true, false><<<gD, 256, 0, stream>>>(
      h_bf, wkt, bk, nullptr, nullptr, k_bf, CMT, CD, CD);
  gemm_kernel<false, false, false, false, true><<<gD, 256, 0, stream>>>(
      h_bf, wvt, bv, nullptr, nullptr, vt, CMT, CD, CD);

  rope_kernel<<<(CMT * 512) / 256, 256, 0, stream>>>(q_bf, k_bf);

  const size_t smem = (size_t)16 * CS * 4 + (size_t)16 * CS * 2 + (size_t)16 * CDK * 2;
  attn_kernel<<<CB * CH * (CS / 16), 128, smem, stream>>>(q_bf, k_bf, vt, o_bf);

  gemm_kernel<false, true, true, false, false><<<gD, 256, 0, stream>>>(
      o_bf, wot, bo, x, x2, nullptr, CMT, CD, CD);

  rmsnorm_kernel<<<CMT, 256, 0, stream>>>(x2, g2, h2_bf, CD);

  gemm_kernel<true, false, false, true, false><<<gF, 256, 0, stream>>>(
      h2_bf, w1t, b1, nullptr, nullptr, a1_bf, CMT, CDFF, CD);

  gemm_kernel<false, true, true, false, false><<<gD, 256, 0, stream>>>(
      a1_bf, w2t, b2, x2, out, nullptr, CMT, CD, CDFF);
}
